// S4D_87179246174434
// MI455X (gfx1250) — compile-verified, matrix operand order checked by reading, not run on hardware
//
#include <hip/hip_runtime.h>
#include <hip/hip_bf16.h>

// S4D layer for MI455X (gfx1250, wave32):
//   1) diagonal-SSM scan (replaces FFT conv) + D-skip + exact GELU -> Yt (f16, [B][L][H])
//   2) fused GEMM (v_wmma_f32_16x16x32_f16) + bias + GLU -> out (f32, [B][H][L])

#define S4D_B  8
#define S4D_H  512
#define S4D_H2 1024
#define S4D_L  4096
#define S4D_N2 32

typedef __attribute__((ext_vector_type(16))) _Float16 v16h;
typedef __attribute__((ext_vector_type(8)))  _Float16 v8h;
typedef __attribute__((ext_vector_type(8)))  float    v8f;

// ---------------------------------------------------------------------------
// Kernel 1: convert W (2H x H, f32 row-major) to f16
// ---------------------------------------------------------------------------
__global__ void s4d_convW(const float* __restrict__ W, _Float16* __restrict__ Wh, int n) {
    int i = blockIdx.x * blockDim.x + threadIdx.x;
    if (i < n) Wh[i] = (_Float16)W[i];
}

// ---------------------------------------------------------------------------
// Kernel 2: SSM scan. One wave per (b,h); lane = complex mode n.
//   x_n[l] = lambda_n * x_n[l-1] + u[l]
//   y[l]   = Re( sum_n 2*Cmod_n * x_n[l] ) + D*u[l]  -> GELU(exact) -> f16
// Stored transposed: Yt[(b*L + l)*H + h] so GEMM K-dim (h) is contiguous.
// ---------------------------------------------------------------------------
__global__ void s4d_scan(const float* __restrict__ u,
                         const float* __restrict__ C,        // (H, N2, 2)
                         const float* __restrict__ log_dt,   // (H)
                         const float* __restrict__ log_A_real,// (H, N2)
                         const float* __restrict__ A_imag,   // (H, N2)
                         const float* __restrict__ Dvec,     // (H)
                         _Float16* __restrict__ Yt)
{
    const int wid  = threadIdx.x >> 5;
    const int lane = threadIdx.x & 31;
    const int gid  = blockIdx.x * (blockDim.x >> 5) + wid;   // 0..B*H-1
    const int b = gid >> 9;          // /H
    const int h = gid & (S4D_H - 1);
    const int n = lane;

    // per-mode constants
    const float dt  = __expf(log_dt[h]) ;
    const float ar  = -__expf(log_A_real[h * S4D_N2 + n]);
    const float ai  = A_imag[h * S4D_N2 + n];
    const float dre = ar * dt, dimg = ai * dt;
    const float er  = __expf(dre);
    const float lam_re = er * __cosf(dimg);
    const float lam_im = er * __sinf(dimg);
    // (exp(dtA) - 1) / A  =  (lam-1) * conj(A) / |A|^2
    const float nr = lam_re - 1.0f, ni = lam_im;
    const float inv_den = 1.0f / (ar * ar + ai * ai);
    const float qr = (nr * ar + ni * ai) * inv_den;
    const float qi = (ni * ar - nr * ai) * inv_den;
    const float c0r = C[(h * S4D_N2 + n) * 2 + 0];
    const float c0i = C[(h * S4D_N2 + n) * 2 + 1];
    const float cr  = 2.0f * (c0r * qr - c0i * qi);
    const float ci  = 2.0f * (c0r * qi + c0i * qr);
    const float Dv  = Dvec[h];

    const float* __restrict__ up = u + ((size_t)b * S4D_H + h) * S4D_L;
    _Float16*   __restrict__ yp = Yt + (size_t)b * S4D_L * S4D_H + h;

    float xr = 0.0f, xi = 0.0f;
    for (int l0 = 0; l0 < S4D_L; l0 += 32) {
        const float uv = up[l0 + lane];          // coalesced 128B per wave
        float csave = 0.0f;
        #pragma unroll
        for (int j = 0; j < 32; ++j) {
            const float uj = __shfl(uv, j, 32);  // broadcast step-j input
            const float t  = lam_re * xr - lam_im * xi + uj;
            xi = lam_re * xi + lam_im * xr;
            xr = t;
            float c = cr * xr - ci * xi;         // Re(2*Cmod * x) per mode
            c += __shfl_xor(c, 16, 32);          // sum the 32 modes
            c += __shfl_xor(c,  8, 32);
            c += __shfl_xor(c,  4, 32);
            c += __shfl_xor(c,  2, 32);
            c += __shfl_xor(c,  1, 32);
            if (lane == j) csave = c;            // lane j keeps y[l0+j]
        }
        // deferred skip + exact GELU (once per lane, not per step)
        const float y = csave + Dv * uv;
        const float g = 0.5f * y * (1.0f + erff(y * 0.70710678118654752f));
        yp[(size_t)(l0 + lane) * S4D_H] = (_Float16)g;   // transposed store
    }
}

// ---------------------------------------------------------------------------
// Kernel 3: fused GEMM + bias + GLU using v_wmma_f32_16x16x32_f16.
//   Z[o, c] = sum_h Wh[o,h] * Yt[c,h],  c = b*L + l
//   out[b, o, l] = (Z[o,c]+bias[o]) * sigmoid(Z[o+512,c]+bias[o+512])
// Each wave: one 16(o) x 16(col) tile, both the 'a' strip and 'g' strip.
// Fragment layout (16-bit, per ISA 7.12.2): lanes 0-15 hold K {k..k+7, k+16..k+23},
// lanes 16-31 hold K {k+8..k+15, k+24..k+31}; A row / B col = lane%16.
// ---------------------------------------------------------------------------
__global__ void s4d_gemm_glu(const _Float16* __restrict__ Wh,   // (2H, H) f16
                             const float*    __restrict__ bias, // (2H)
                             const _Float16* __restrict__ Yt,   // (B*L, H) f16
                             float*          __restrict__ out)  // (B, H, L)
{
    const int wid  = threadIdx.x >> 5;
    const int lane = threadIdx.x & 31;
    const int lrow = lane & 15;
    const int hi   = lane >> 4;
    const int koff = hi << 3;                    // 0 or 8

    const int colTile = blockIdx.x * 8 + wid;    // 0..2047
    const int c0 = colTile << 4;
    const int b  = c0 >> 12;                     // /L
    const int l0 = c0 & (S4D_L - 1);
    const int o0 = blockIdx.y << 4;              // 0..511 step 16

    const _Float16* __restrict__ wa = Wh + (size_t)(o0 + lrow) * S4D_H;
    const _Float16* __restrict__ wg = Wh + (size_t)(o0 + S4D_H + lrow) * S4D_H;
    const _Float16* __restrict__ yb = Yt + ((size_t)b * S4D_L + l0 + lrow) * S4D_H;

    v8f accA = {};
    v8f accG = {};
    union Frag { v16h v; v8h h[2]; };

    #pragma unroll
    for (int k = 0; k < S4D_H; k += 32) {
        Frag fa, fg, fb;
        fa.h[0] = *(const v8h*)(wa + k + koff);
        fa.h[1] = *(const v8h*)(wa + k + 16 + koff);
        fg.h[0] = *(const v8h*)(wg + k + koff);
        fg.h[1] = *(const v8h*)(wg + k + 16 + koff);
        fb.h[0] = *(const v8h*)(yb + k + koff);
        fb.h[1] = *(const v8h*)(yb + k + 16 + koff);
        accA = __builtin_amdgcn_wmma_f32_16x16x32_f16(false, fa.v, false, fb.v,
                                                      (short)0, accA, false, false);
        accG = __builtin_amdgcn_wmma_f32_16x16x32_f16(false, fg.v, false, fb.v,
                                                      (short)0, accG, false, false);
    }

    // C/D layout: VGPR r -> row o0 + r + 8*hi, col = c0 + lrow
    const int rbase = hi << 3;
    #pragma unroll
    for (int r = 0; r < 8; ++r) {
        const int o = o0 + rbase + r;
        const float av = accA[r] + bias[o];
        const float gv = accG[r] + bias[o + S4D_H];
        const float sg = 1.0f / (1.0f + __expf(-gv));
        out[((size_t)b * S4D_H + o) * S4D_L + l0 + lrow] = av * sg;
    }
}

// ---------------------------------------------------------------------------
extern "C" void kernel_launch(void* const* d_in, const int* in_sizes, int n_in,
                              void* d_out, int out_size, void* d_ws, size_t ws_size,
                              hipStream_t stream) {
    const float* u          = (const float*)d_in[0];
    const float* C          = (const float*)d_in[1];
    const float* log_dt     = (const float*)d_in[2];
    const float* log_A_real = (const float*)d_in[3];
    const float* A_imag     = (const float*)d_in[4];
    const float* D          = (const float*)d_in[5];
    const float* W          = (const float*)d_in[6];
    const float* bias       = (const float*)d_in[7];
    float* out = (float*)d_out;

    // workspace layout: [ Wh f16 : 1MB ][ Yt f16 : 32MB ]
    _Float16* Wh = (_Float16*)d_ws;
    _Float16* Yt = (_Float16*)((char*)d_ws + (size_t)S4D_H2 * S4D_H * sizeof(_Float16));

    // 1) W -> f16
    {
        const int n = S4D_H2 * S4D_H;
        s4d_convW<<<(n + 255) / 256, 256, 0, stream>>>(W, Wh, n);
    }
    // 2) SSM scan + GELU -> Yt (one wave per (b,h); 8 waves/block)
    {
        const int nwaves = S4D_B * S4D_H;          // 4096
        s4d_scan<<<nwaves / 8, 256, 0, stream>>>(u, C, log_dt, log_A_real,
                                                 A_imag, D, Yt);
    }
    // 3) WMMA GEMM + GLU -> out
    {
        dim3 grid((S4D_B * S4D_L / 16) / 8, S4D_H / 16);   // (256, 32)
        s4d_gemm_glu<<<grid, 256, 0, stream>>>(Wh, bias, Yt, out);
    }
}
